// Message_10187662426871
// MI455X (gfx1250) — matrix operand order repaired and checked by reading, not compiled, not run on hardware
//
#include <hip/hip_runtime.h>
#include <hip/hip_bf16.h>

typedef __attribute__((ext_vector_type(16))) _Float16 v16h;
typedef __attribute__((ext_vector_type(8)))  _Float16 v8h;
typedef __attribute__((ext_vector_type(8)))  float    v8f;
typedef __attribute__((ext_vector_type(4)))  unsigned u32x4;
typedef __attribute__((ext_vector_type(8)))  unsigned u32x8;

#define NF    128
#define NNODE 10000
#define NEDGE 160000
#define EQ_ELEMS (NNODE * NF * 3)   // 3,840,000
#define IN_ELEMS (NNODE * NF)       // 1,280,000
#define WPB   4                      // waves per block in edge kernel

// ---------------------------------------------------------------------------
// WMMA fragment helpers (wave32, 16x16x32 f16 -> f32)
// A 16x32 f16 layout (ISA 7.12.2): lane L holds row M = L&15;
//   halves [0..7]  = K = kk + hi*8 .. +7,   halves [8..15] = K = kk+16+hi*8 .. +7
// ---------------------------------------------------------------------------
template <typename P>
__device__ __forceinline__ v16h frag_a(P base, int ldk, int kk, int lane) {
    const int r  = lane & 15;
    const int hi = (lane >> 4) & 1;
    P p0 = base + (size_t)r * ldk + kk + hi * 8;
    v8h lo = *(const v8h*)p0;
    v8h hh = *(const v8h*)(p0 + 16);
    v16h out;
#pragma unroll
    for (int i = 0; i < 8; ++i) { out[i] = lo[i]; out[i + 8] = hh[i]; }
    return out;
}

// B 32x16 f16 (K x N), weights stored transposed as Wt[N][K]:
// lane L holds column n0 + (L&15); halves [0..7] = K = kk + hi*8 .. +7, [8..15] = K = kk+16+hi*8 .. +7.
template <typename P>
__device__ __forceinline__ v16h frag_b(P wt, int ldk, int n0, int kk, int lane) {
    const int n  = n0 + (lane & 15);
    const int kb = kk + ((lane >> 4) & 1) * 8;
    P p = wt + (size_t)n * ldk + kb;
    v8h lo = *(const v8h*)p;
    v8h hh = *(const v8h*)(p + 16);
    v16h out;
#pragma unroll
    for (int i = 0; i < 8; ++i) { out[i] = lo[i]; out[i + 8] = hh[i]; }
    return out;
}

__device__ __forceinline__ v8f wmma_f16(v16h a, v16h b, v8f c) {
    return __builtin_amdgcn_wmma_f32_16x16x32_f16(false, a, false, b, (short)0, c,
                                                  false, false);
}

__device__ __forceinline__ float silu(float x) {
    return x / (1.0f + __expf(-x));
}

// ---------------------------------------------------------------------------
// Prep kernels
// ---------------------------------------------------------------------------
__global__ void k_transpose_f16(const float* __restrict__ w, _Float16* __restrict__ wt,
                                int K, int N) {
    int idx = blockIdx.x * blockDim.x + threadIdx.x;   // idx = n*K + k
    if (idx < K * N) {
        int n = idx / K, k = idx % K;
        wt[idx] = (_Float16)w[(size_t)k * N + n];
    }
}

__global__ void k_to_f16(const float* __restrict__ src, _Float16* __restrict__ dst, int n) {
    int idx = blockIdx.x * blockDim.x + threadIdx.x;
    if (idx < n) dst[idx] = (_Float16)src[idx];
}

// ---------------------------------------------------------------------------
// Node phi MLP, stage 1: H = silu(inv @ W1 + b1)   (10000x128)
// ---------------------------------------------------------------------------
__global__ __launch_bounds__(128) void k_node_g1(const _Float16* __restrict__ inv16,
                                                 const _Float16* __restrict__ wt1,
                                                 const float* __restrict__ b1,
                                                 _Float16* __restrict__ hout) {
    const int lane = threadIdx.x & 31;
    const int tile = blockIdx.x * 4 + (threadIdx.x >> 5);
    if (tile >= NNODE / 16) return;
    const int nbase = tile * 16;
    const int nl = lane & 15, hi = (lane >> 4) & 1;

    v16h af[4];
#pragma unroll
    for (int kk = 0; kk < 4; ++kk)
        af[kk] = frag_a(inv16 + (size_t)nbase * NF, NF, kk * 32, lane);

#pragma unroll
    for (int nt = 0; nt < 8; ++nt) {
        v8f c = {};
#pragma unroll
        for (int kk = 0; kk < 4; ++kk)
            c = wmma_f16(af[kk], frag_b(wt1, NF, nt * 16, kk * 32, lane), c);
        const float bv = b1[nt * 16 + nl];
#pragma unroll
        for (int r = 0; r < 8; ++r) {
            float h = silu(c[r] + bv);
            hout[(size_t)(nbase + r + 8 * hi) * NF + nt * 16 + nl] = (_Float16)h;
        }
    }
}

// ---------------------------------------------------------------------------
// Node phi MLP, stage 2: Mphi = H @ W2 + b2   (10000x512 f32)
// ---------------------------------------------------------------------------
__global__ __launch_bounds__(128) void k_node_g2(const _Float16* __restrict__ hphi,
                                                 const _Float16* __restrict__ wt2,
                                                 const float* __restrict__ b2,
                                                 float* __restrict__ mphi) {
    const int lane = threadIdx.x & 31;
    const int wid  = blockIdx.x * 4 + (threadIdx.x >> 5);
    if (wid >= (NNODE / 16) * 8) return;
    const int tile = wid >> 3, chunk = wid & 7;
    const int nbase = tile * 16;
    const int nl = lane & 15, hi = (lane >> 4) & 1;

    v16h af[4];
#pragma unroll
    for (int kk = 0; kk < 4; ++kk)
        af[kk] = frag_a(hphi + (size_t)nbase * NF, NF, kk * 32, lane);

#pragma unroll
    for (int t = 0; t < 4; ++t) {
        const int n0 = chunk * 64 + t * 16;
        v8f c = {};
#pragma unroll
        for (int kk = 0; kk < 4; ++kk)
            c = wmma_f16(af[kk], frag_b(wt2, NF, n0, kk * 32, lane), c);
        const float bv = b2[n0 + nl];
#pragma unroll
        for (int r = 0; r < 8; ++r)
            mphi[(size_t)(nbase + r + 8 * hi) * 512 + n0 + nl] = c[r] + bv;
    }
}

// ---------------------------------------------------------------------------
// Edge kernel: pe -> mlp_w (WMMA) -> gate with mphi[src] -> cross/scatter
// One wave per 16-edge tile, WPB waves per block.
// w_w1 (128x128 f16, transposed) is staged to LDS once per block via the
// Tensor Data Mover (tensor_load_to_lds + s_wait_tensorcnt), then all GEMM1
// B-fragments are ds_load_b128 reads.
// ---------------------------------------------------------------------------
__global__ __launch_bounds__(WPB * 32) void k_edges(
    const int* __restrict__ ei, const float* __restrict__ dist,
    const float* __restrict__ dir, const float* __restrict__ inv,
    const float* __restrict__ equiv, const _Float16* __restrict__ wt_w1,
    const float* __restrict__ wb1, const _Float16* __restrict__ wt_w2,
    const float* __restrict__ wb2, const float* __restrict__ mphi,
    float* __restrict__ out_eq, float* __restrict__ out_in) {
    __shared__ _Float16 s_w1[NF * NF];          // 32 KB staged weights (N x K)
    __shared__ _Float16 s_h[WPB][16][136];      // per-wave Hw staging
    __shared__ int   s_src[WPB][16], s_dst[WPB][16];
    __shared__ float s_dir[WPB][16][3];

    const int lane = threadIdx.x & 31;
    const int wv   = threadIdx.x >> 5;
    const int tile = blockIdx.x * WPB + wv;
    const int nl = lane & 15, hi = (lane >> 4) & 1;

    // ---- TDM: stage w_w1^T into LDS (wave 0 issues; barrier publishes) ----
    if (wv == 0) {
        const unsigned lds_off = (unsigned)(unsigned long long)&s_w1[0];
        const unsigned long long ga = (unsigned long long)wt_w1;
        u32x4 g0;
        g0[0] = 1u;                                   // count=1 (valid user D#)
        g0[1] = lds_off;                              // lds_addr
        g0[2] = (unsigned)ga;                         // global_addr[31:0]
        g0[3] = (unsigned)(ga >> 32) | (2u << 30);    // global_addr hi | type=2
        u32x8 g1;
        g1[0] = 1u << 16;          // data_size=1 (2 bytes), no mask/flags
        g1[1] = 16384u << 16;      // tensor_dim0[15:0] at bits [63:48]
        g1[2] = 1u << 16;          // tensor_dim0 hi=0 | tensor_dim1[15:0]=1
        g1[3] = 16384u << 16;      // tensor_dim1 hi=0 | tile_dim0=16384
        g1[4] = 1u;                // tile_dim1=1, tile_dim2=0
        g1[5] = 16384u;            // tensor_dim0_stride[31:0]
        g1[6] = 16384u << 16;      // stride0 hi=0 | tensor_dim1_stride[15:0]
        g1[7] = 0u;                // tensor_dim1_stride hi
        asm volatile("tensor_load_to_lds %0, %1" :: "s"(g0), "s"(g1) : "memory");
        __builtin_amdgcn_s_wait_tensorcnt(0);
    }
    __syncthreads();
    if (tile >= NEDGE / 16) return;
    const int ebase = tile * 16;

    if (lane < 16) {
        int e = ebase + lane;
        s_src[wv][lane] = ei[e];
        s_dst[wv][lane] = ei[NEDGE + e];
        s_dir[wv][lane][0] = dir[3 * e + 0];
        s_dir[wv][lane][1] = dir[3 * e + 1];
        s_dir[wv][lane][2] = dir[3 * e + 2];
    }

    // --- generate positional-encoding A fragments (row = edge nl) ---
    const float d  = dist[ebase + nl];
    const float w0 = 6.28318530717958647692f / 10.0f;   // 2*pi / length_scale
    v16h af[4];
#pragma unroll
    for (int kk = 0; kk < 4; ++kk) {
        v16h o;
#pragma unroll
        for (int half = 0; half < 2; ++half) {
            const int kb = kk * 32 + half * 16 + hi * 8;
#pragma unroll
            for (int i = 0; i < 8; i += 2) {
                float rank = (float)((kb + i) / 2 + 1);
                float s, c;
                __sincosf(w0 * d * rank, &s, &c);
                o[half * 8 + i]     = (_Float16)s;
                o[half * 8 + i + 1] = (_Float16)c;
            }
        }
        af[kk] = o;
    }

    // --- GEMM1: Hw = silu(pe @ w_w1 + b1); B-frags from LDS-staged weights ---
#pragma unroll
    for (int nt = 0; nt < 8; ++nt) {
        v8f c = {};
#pragma unroll
        for (int kk = 0; kk < 4; ++kk)
            c = wmma_f16(af[kk], frag_b((const _Float16*)&s_w1[0], NF, nt * 16,
                                        kk * 32, lane), c);
        const float bv = wb1[nt * 16 + nl];
#pragma unroll
        for (int r = 0; r < 8; ++r)
            s_h[wv][r + 8 * hi][nt * 16 + nl] = (_Float16)silu(c[r] + bv);
    }

    // --- A fragments for GEMM2 from LDS-staged Hw ---
    v16h a2[4];
#pragma unroll
    for (int kk = 0; kk < 4; ++kk)
        a2[kk] = frag_a((const _Float16*)&s_h[wv][0][0], 136, kk * 32, lane);

    // --- vector gates: groups 0 (gate), 1 (cross), 2 (edge_dir scale) ---
#pragma unroll
    for (int cc = 0; cc < 2; ++cc) {
#pragma unroll
        for (int t = 0; t < 4; ++t) {
            const int n0 = cc * 64 + t * 16;
            v8f cg = {}, cx = {}, ce = {};
#pragma unroll
            for (int kk = 0; kk < 4; ++kk) {
                cg = wmma_f16(a2[kk], frag_b(wt_w2, NF, n0,        kk * 32, lane), cg);
                cx = wmma_f16(a2[kk], frag_b(wt_w2, NF, 128 + n0,  kk * 32, lane), cx);
                ce = wmma_f16(a2[kk], frag_b(wt_w2, NF, 256 + n0,  kk * 32, lane), ce);
            }
            const int feat = n0 + nl;
            const float bg = wb2[feat], bc = wb2[128 + feat], be = wb2[256 + feat];
#pragma unroll
            for (int r = 0; r < 8; ++r) {
                const int er = r + 8 * hi;
                const int sn = s_src[wv][er], dn = s_dst[wv][er];
                const float dx = s_dir[wv][er][0], dy = s_dir[wv][er][1],
                            dz = s_dir[wv][er][2];
                const float* mp = mphi + (size_t)sn * 512;
                const float mg = mp[feat]        * (cg[r] + bg);
                const float mc = mp[128 + feat]  * (cx[r] + bc);
                const float me = mp[256 + feat]  * (ce[r] + be);
                const float* es = equiv + ((size_t)sn * NF + feat) * 3;
                const float* ed = equiv + ((size_t)dn * NF + feat) * 3;
                const float ex = es[0], ey = es[1], ez = es[2];
                const float vx = ed[0], vy = ed[1], vz = ed[2];
                const float crx = dy * vz - dz * vy;
                const float cry = dz * vx - dx * vz;
                const float crz = dx * vy - dy * vx;
                float* o = out_eq + ((size_t)dn * NF + feat) * 3;
                atomicAdd(o + 0, me * dx + mg * ex + mc * crx);
                atomicAdd(o + 1, me * dy + mg * ey + mc * cry);
                atomicAdd(o + 2, me * dz + mg * ez + mc * crz);
            }
        }
    }

    // --- scalar gate: group 3 (scale_features * inv[src]) ---
#pragma unroll
    for (int cc = 0; cc < 2; ++cc) {
#pragma unroll
        for (int t = 0; t < 4; ++t) {
            const int n0 = cc * 64 + t * 16;
            v8f cs = {};
#pragma unroll
            for (int kk = 0; kk < 4; ++kk)
                cs = wmma_f16(a2[kk], frag_b(wt_w2, NF, 384 + n0, kk * 32, lane), cs);
            const int feat = n0 + nl;
            const float bs = wb2[384 + feat];
#pragma unroll
            for (int r = 0; r < 8; ++r) {
                const int er = r + 8 * hi;
                const int sn = s_src[wv][er], dn = s_dst[wv][er];
                const float dsv = mphi[(size_t)sn * 512 + 384 + feat] * (cs[r] + bs) *
                                  inv[(size_t)sn * NF + feat];
                atomicAdd(&out_in[(size_t)dn * NF + feat], dsv);
            }
        }
    }
}

// ---------------------------------------------------------------------------
extern "C" void kernel_launch(void* const* d_in, const int* in_sizes, int n_in,
                              void* d_out, int out_size, void* d_ws, size_t ws_size,
                              hipStream_t stream) {
    (void)in_sizes; (void)n_in; (void)out_size; (void)ws_size;
    const int*   edge_index = (const int*)d_in[0];
    const float* edge_dist  = (const float*)d_in[1];
    const float* edge_dir   = (const float*)d_in[2];
    const float* inv        = (const float*)d_in[3];
    const float* equiv      = (const float*)d_in[4];
    const float* phi_w1 = (const float*)d_in[5];
    const float* phi_b1 = (const float*)d_in[6];
    const float* phi_w2 = (const float*)d_in[7];
    const float* phi_b2 = (const float*)d_in[8];
    const float* w_w1   = (const float*)d_in[9];
    const float* w_b1   = (const float*)d_in[10];
    const float* w_w2   = (const float*)d_in[11];
    const float* w_b2   = (const float*)d_in[12];

    float* out_eq = (float*)d_out;              // [10000,128,3]
    float* out_in = (float*)d_out + EQ_ELEMS;   // [10000,128]

    char* ws = (char*)d_ws;
    size_t off = 0;
    auto alloc = [&](size_t bytes) -> void* {
        void* p = ws + off;
        off = (off + bytes + 255) & ~(size_t)255;
        return p;
    };
    _Float16* wt_phi1 = (_Float16*)alloc(128 * 128 * 2);
    _Float16* wt_phi2 = (_Float16*)alloc(512 * 128 * 2);
    _Float16* wt_w1   = (_Float16*)alloc(128 * 128 * 2);
    _Float16* wt_w2   = (_Float16*)alloc(512 * 128 * 2);
    _Float16* inv16   = (_Float16*)alloc((size_t)NNODE * NF * 2);
    _Float16* hphi    = (_Float16*)alloc((size_t)NNODE * NF * 2);
    float*    mphi    = (float*)alloc((size_t)NNODE * 512 * 4);

    // outputs start as inputs (reference adds aggregates on top)
    hipMemcpyAsync(out_eq, equiv, (size_t)EQ_ELEMS * 4, hipMemcpyDeviceToDevice, stream);
    hipMemcpyAsync(out_in, inv,   (size_t)IN_ELEMS * 4, hipMemcpyDeviceToDevice, stream);

    // weight transpose + f16 conversion
    k_transpose_f16<<<(128 * 128 + 255) / 256, 256, 0, stream>>>(phi_w1, wt_phi1, 128, 128);
    k_transpose_f16<<<(512 * 128 + 255) / 256, 256, 0, stream>>>(phi_w2, wt_phi2, 128, 512);
    k_transpose_f16<<<(128 * 128 + 255) / 256, 256, 0, stream>>>(w_w1, wt_w1, 128, 128);
    k_transpose_f16<<<(512 * 128 + 255) / 256, 256, 0, stream>>>(w_w2, wt_w2, 128, 512);
    k_to_f16<<<(IN_ELEMS + 255) / 256, 256, 0, stream>>>(inv, inv16, IN_ELEMS);

    // per-node phi MLP (16x fewer FLOPs than per-edge)
    k_node_g1<<<(NNODE / 16 + 3) / 4, 128, 0, stream>>>(inv16, wt_phi1, phi_b1, hphi);
    k_node_g2<<<((NNODE / 16) * 8 + 3) / 4, 128, 0, stream>>>(hphi, wt_phi2, phi_b2, mphi);

    // per-edge: pe -> mlp_w (WMMA) -> fused gating + scatter
    k_edges<<<(NEDGE / 16) / WPB, WPB * 32, 0, stream>>>(
        edge_index, edge_dist, edge_dir, inv, equiv, wt_w1, w_b1, wt_w2, w_b2, mphi,
        out_eq, out_in);
}
